// MaskedMultiHeadAttention_9174050144525
// MI455X (gfx1250) — matrix operand order repaired and checked by reading, not compiled, not run on hardware
//
#include <hip/hip_runtime.h>
#include <hip/hip_bf16.h>
#include <math.h>

typedef __attribute__((ext_vector_type(16))) __bf16        v16bf;
typedef __attribute__((ext_vector_type(8)))  float         v8f;
typedef __attribute__((ext_vector_type(4)))  unsigned int  u32x4;

#define EMBED 1024
#define HEADS 16
#define HD    64
#define BB    2
#define SEQ   2048
#define MROWS (BB * SEQ) /* 4096 */

static __device__ __forceinline__ __bf16 f2bf(float f) {
  union { float f; unsigned int u; } v; v.f = f;
  unsigned int r = (v.u + 0x7FFFu + ((v.u >> 16) & 1u)) >> 16;
  union { unsigned short s; __bf16 b; } o; o.s = (unsigned short)r;
  return o.b;
}

// A fragment, 16x32 bf16, row-major source (ISA 7.12.2 16-bit A layout):
// lane (half,m): VGPR j<4: K = half*8 + 2j (+0/1); VGPR 4+j: K = 16 + half*8 + 2j
// -> two contiguous 8-element runs per lane -> 2x ds_load_b128
static __device__ __forceinline__ v16bf load_a_frag(const __bf16* p, int ld, int lane) {
  const int lh = lane >> 4, m = lane & 15;
  const __bf16* row = p + m * ld;
  v16bf r;
#pragma unroll
  for (int j = 0; j < 4; ++j) {
    int k0 = lh * 8 + 2 * j;
    r[2*j]   = row[k0];
    r[2*j+1] = row[k0+1];
    int k1 = 16 + lh * 8 + 2 * j;
    r[8+2*j] = row[k1];
    r[9+2*j] = row[k1+1];
  }
  return r;
}

// B fragment from *transposed* storage Bt[n][k] (row-major N x Kdepth):
// lane (half,n) holds K = half*8 + 0..7 and 16 + half*8 + 0..7 at column n
// -> 2x ds_load_b128 per lane. For X @ W^T, Bt == W row-major (no transpose needed).
static __device__ __forceinline__ v16bf load_bt_frag(const __bf16* p, int ld, int lane) {
  const int lh = lane >> 4, n = lane & 15;
  const __bf16* row = p + n * ld + lh * 8;
  v16bf r;
#pragma unroll
  for (int e = 0; e < 16; ++e) r[e] = row[e < 8 ? e : 8 + e];
  return r;
}

__global__ __launch_bounds__(256) void cvt_f32_to_bf16(const float* __restrict__ in,
                                                       __bf16* __restrict__ out, int n) {
  int i = blockIdx.x * 256 + threadIdx.x;
  if (i < n) out[i] = f2bf(in[i]);
}

// C = A(bf16 MxK row-major) @ W^T + bias, W staged row-major (N x K).
// Double-buffered LDS, 64-wide K steps, global loads pipelined behind WMMA.
// mode 0: bf16 scatter to [B, H, S, HD]; mode 1: f32 row-major [M, N]
__global__ __launch_bounds__(256) void gemm_bf16_128x128(
    const __bf16* __restrict__ A, const __bf16* __restrict__ Wb,
    const float* __restrict__ bias, __bf16* __restrict__ outb,
    float* __restrict__ outf, int M, int N, int K, int mode) {
  __shared__ alignas(16) __bf16 Atile[2][128 * 64];   // [m][k]
  __shared__ alignas(16) __bf16 Btile[2][128 * 64];   // [n][k]  (W rows)
  const int tid = threadIdx.x;
  const int lane = tid & 31, wid = tid >> 5;
  const int wave_m = wid & 3, wave_n = wid >> 2;      // 4 x 2 wave grid
  const int bm = blockIdx.y * 128, bn = blockIdx.x * 128;
  const int lh = lane >> 4, nn = lane & 15;

  v8f acc[2][4];
#pragma unroll
  for (int i = 0; i < 2; ++i)
#pragma unroll
    for (int j = 0; j < 4; ++j)
#pragma unroll
      for (int e = 0; e < 8; ++e) acc[i][j][e] = 0.f;

  const int nk = K >> 6;
  u32x4 ra[4], rb[4];

  // prologue: fetch K-tile 0 into regs, stage into buffer 0
#pragma unroll
  for (int i = 0; i < 4; ++i) {
    int idx = tid + i * 256;
    int row = idx >> 3, ch = idx & 7;
    ra[i] = *(const u32x4*)&A[(size_t)(bm + row) * K + ch * 8];
    rb[i] = *(const u32x4*)&Wb[(size_t)(bn + row) * K + ch * 8];
  }
#pragma unroll
  for (int i = 0; i < 4; ++i) {
    int idx = tid + i * 256;
    int row = idx >> 3, ch = idx & 7;
    *(u32x4*)&Atile[0][row * 64 + ch * 8] = ra[i];
    *(u32x4*)&Btile[0][row * 64 + ch * 8] = rb[i];
  }

  for (int kt = 0; kt < nk; ++kt) {
    __syncthreads();                                  // stage visible / prev reads done
    const int buf = kt & 1;
    if (kt + 1 < nk) {                                // issue next tile's loads early
      const int kk = (kt + 1) << 6;
#pragma unroll
      for (int i = 0; i < 4; ++i) {
        int idx = tid + i * 256;
        int row = idx >> 3, ch = idx & 7;
        ra[i] = *(const u32x4*)&A[(size_t)(bm + row) * K + kk + ch * 8];
        rb[i] = *(const u32x4*)&Wb[(size_t)(bn + row) * K + kk + ch * 8];
      }
    }
#pragma unroll
    for (int s2 = 0; s2 < 2; ++s2) {                  // two 32-deep WMMA steps
      const int kk2 = s2 * 32;
      v16bf af[2], bfm[4];
#pragma unroll
      for (int mi = 0; mi < 2; ++mi)
        af[mi] = load_a_frag(&Atile[buf][(wave_m * 32 + mi * 16) * 64 + kk2], 64, lane);
#pragma unroll
      for (int ni = 0; ni < 4; ++ni)
        bfm[ni] = load_bt_frag(&Btile[buf][(wave_n * 64 + ni * 16) * 64 + kk2], 64, lane);
#pragma unroll
      for (int mi = 0; mi < 2; ++mi)
#pragma unroll
        for (int ni = 0; ni < 4; ++ni)
          acc[mi][ni] = __builtin_amdgcn_wmma_f32_16x16x32_bf16(
              false, af[mi], false, bfm[ni], (short)0, acc[mi][ni], false, false);
    }
    if (kt + 1 < nk) {                                // stage next tile
#pragma unroll
      for (int i = 0; i < 4; ++i) {
        int idx = tid + i * 256;
        int row = idx >> 3, ch = idx & 7;
        *(u32x4*)&Atile[buf ^ 1][row * 64 + ch * 8] = ra[i];
        *(u32x4*)&Btile[buf ^ 1][row * 64 + ch * 8] = rb[i];
      }
    }
  }

#pragma unroll
  for (int mi = 0; mi < 2; ++mi)
#pragma unroll
    for (int ni = 0; ni < 4; ++ni)
#pragma unroll
      for (int r = 0; r < 8; ++r) {
        int m = bm + wave_m * 32 + mi * 16 + lh * 8 + r;
        int n = bn + wave_n * 64 + ni * 16 + nn;
        float v = acc[mi][ni][r] + bias[n];
        if (mode == 0) {
          int b = m >> 11, s = m & (SEQ - 1);
          int h = n >> 6, d = n & (HD - 1);
          outb[(((size_t)(b * HEADS + h)) * SEQ + s) * HD + d] = f2bf(v);
        } else {
          outf[(size_t)m * N + n] = v;
        }
      }
}

// Flash attention: per (b, h, 64-row q tile). Key tiles of 64 streamed via LDS.
__global__ __launch_bounds__(256) void attn_fwd(
    const __bf16* __restrict__ Qh, const __bf16* __restrict__ Kh,
    const __bf16* __restrict__ Vh, const int* __restrict__ pm,
    __bf16* __restrict__ Ctx) {
  __shared__ alignas(16) __bf16 Qs[64 * 64];    // [q][d]
  __shared__ alignas(16) __bf16 Ks[64 * 64];    // [s_local][d] (row-major == Bt for QK^T)
  __shared__ alignas(16) __bf16 Vt[64 * 64];    // [d][s_local] (Bt for P@V)
  __shared__ alignas(16) float  Ssc[64 * 64];   // f32 scores; bf16 probs aliased in-place
  __shared__ float mrow[64], lrow[64], arow[64];
  __shared__ int kmaskS[64];
  __bf16* Ps = (__bf16*)Ssc;                    // P(r,c) = Ps[r*128 + c]

  const int tid = threadIdx.x;
  const int lane = tid & 31, wid = tid >> 5;
  const int wave_m = wid & 3;                   // q 16-row tile
  const int wn2 = (wid >> 2) * 2;               // first of two 16-col tiles
  const int lh = lane >> 4, nn = lane & 15;

  const int qt = blockIdx.x, h = blockIdx.y, b = blockIdx.z;
  const int q0 = qt * 64;
  const size_t headBase = ((size_t)(b * HEADS + h)) * SEQ * HD;

#pragma unroll
  for (int i = 0; i < 2; ++i) {                 // load 64x64 Q tile
    int idx = tid + i * 256;
    int row = idx >> 3, ch = idx & 7;
    *(u32x4*)&Qs[row * 64 + ch * 8] =
        *(const u32x4*)&Qh[headBase + (size_t)(q0 + row) * HD + ch * 8];
  }
  if (tid < 64) { mrow[tid] = -INFINITY; lrow[tid] = 0.f; }
  const int qmask = pm[b * SEQ + q0 + (tid & 63)];

  v8f Oacc[2];
#pragma unroll
  for (int f = 0; f < 2; ++f)
#pragma unroll
    for (int e = 0; e < 8; ++e) Oacc[f][e] = 0.f;
  __syncthreads();

  for (int kt = 0; kt < SEQ / 64; ++kt) {
    const int k0 = kt * 64;
    u32x4 rk[2], rv[2];                         // fetch K/V/mask before the barrier:
    int km = 0;                                 // global latency overlaps prev PV tail
#pragma unroll
    for (int i = 0; i < 2; ++i) {
      int idx = tid + i * 256;
      int row = idx >> 3, ch = idx & 7;
      rk[i] = *(const u32x4*)&Kh[headBase + (size_t)(k0 + row) * HD + ch * 8];
      rv[i] = *(const u32x4*)&Vh[headBase + (size_t)(k0 + row) * HD + ch * 8];
    }
    if (tid < 64) km = pm[b * SEQ + k0 + tid];
    __syncthreads();                            // previous tile's LDS reads done
#pragma unroll
    for (int i = 0; i < 2; ++i) {
      int idx = tid + i * 256;
      int row = idx >> 3, ch = idx & 7;
      *(u32x4*)&Ks[row * 64 + ch * 8] = rk[i];  // K row-major
      union { u32x4 v; __bf16 hv[8]; } u; u.v = rv[i];
#pragma unroll
      for (int e = 0; e < 8; ++e)               // V transposed -> [d][s]
        Vt[(ch * 8 + e) * 64 + row] = u.hv[e];
    }
    if (tid < 64) kmaskS[tid] = km;
    __syncthreads();

    v8f sc[2];                                  // scores = Q @ K^T
#pragma unroll
    for (int f = 0; f < 2; ++f)
#pragma unroll
      for (int e = 0; e < 8; ++e) sc[f][e] = 0.f;
#pragma unroll
    for (int kk = 0; kk < 64; kk += 32) {
      v16bf qa = load_a_frag(&Qs[(wave_m * 16) * 64 + kk], 64, lane);
#pragma unroll
      for (int f = 0; f < 2; ++f) {
        v16bf kb = load_bt_frag(&Ks[((wn2 + f) * 16) * 64 + kk], 64, lane);
        sc[f] = __builtin_amdgcn_wmma_f32_16x16x32_bf16(
            false, qa, false, kb, (short)0, sc[f], false, false);
      }
    }
#pragma unroll
    for (int f = 0; f < 2; ++f)
#pragma unroll
      for (int r = 0; r < 8; ++r)
        Ssc[(wave_m * 16 + lh * 8 + r) * 64 + (wn2 + f) * 16 + nn] = sc[f][r];
    __syncthreads();

    if (tid < 64) {                             // online softmax, 1 thread / q row
      int r = tid;
      float* srow = &Ssc[r * 64];
      __bf16* prow = &Ps[r * 128];
      float alpha = 1.f;
      if (qmask != 0) {
        float mold = mrow[r];
        float mx = mold;
        for (int c = 0; c < 64; ++c)
          if (kmaskS[c]) mx = fmaxf(mx, srow[c] * 0.125f);
        if (mx != -INFINITY) {
          alpha = expf(mold - mx);
          float sum = 0.f;
          for (int c = 0; c < 64; ++c) {        // in-place f32->bf16 is byte-safe
            float p = 0.f;
            if (kmaskS[c]) { p = expf(srow[c] * 0.125f - mx); sum += p; }
            prow[c] = f2bf(p);
          }
          mrow[r] = mx;
          lrow[r] = lrow[r] * alpha + sum;
        } else {
          for (int c = 0; c < 64; ++c) prow[c] = f2bf(0.f);
        }
      } else {
        for (int c = 0; c < 64; ++c) prow[c] = f2bf(0.f);
      }
      arow[r] = alpha;
    }
    __syncthreads();

#pragma unroll
    for (int f = 0; f < 2; ++f)                 // rescale running O
#pragma unroll
      for (int r = 0; r < 8; ++r)
        Oacc[f][r] *= arow[wave_m * 16 + lh * 8 + r];
#pragma unroll
    for (int kk = 0; kk < 64; kk += 32) {       // O += P @ V
      v16bf pa = load_a_frag(&Ps[(wave_m * 16) * 128 + kk], 128, lane);
#pragma unroll
      for (int f = 0; f < 2; ++f) {
        v16bf vb = load_bt_frag(&Vt[((wn2 + f) * 16) * 64 + kk], 64, lane);
        Oacc[f] = __builtin_amdgcn_wmma_f32_16x16x32_bf16(
            false, pa, false, vb, (short)0, Oacc[f], false, false);
      }
    }
  }
  __syncthreads();

#pragma unroll
  for (int f = 0; f < 2; ++f)                   // normalize, store ctx [B,S,EMBED]
#pragma unroll
    for (int r = 0; r < 8; ++r) {
      int row = wave_m * 16 + lh * 8 + r;
      float l = lrow[row];
      float denom = (l == 0.f) ? 1.f : l;
      int gq = q0 + row;
      int col = h * HD + (wn2 + f) * 16 + nn;
      Ctx[((size_t)b * SEQ + gq) * EMBED + col] = f2bf(Oacc[f][r] / denom);
    }
}

extern "C" void kernel_launch(void* const* d_in, const int* in_sizes, int n_in,
                              void* d_out, int out_size, void* d_ws, size_t ws_size,
                              hipStream_t stream) {
  (void)in_sizes; (void)n_in; (void)out_size; (void)ws_size;
  const float* q  = (const float*)d_in[0];
  const float* k  = (const float*)d_in[1];
  const float* v  = (const float*)d_in[2];
  const int*   pm = (const int*)d_in[3];
  const float* Wq = (const float*)d_in[4];
  const float* bq = (const float*)d_in[5];
  const float* Wk = (const float*)d_in[6];
  const float* bk = (const float*)d_in[7];
  const float* Wv = (const float*)d_in[8];
  const float* bv = (const float*)d_in[9];
  const float* Wo = (const float*)d_in[10];
  const float* bo = (const float*)d_in[11];
  float* out = (float*)d_out;

  const size_t SZX = (size_t)MROWS * EMBED;  // 4M bf16
  const size_t SZW = (size_t)EMBED * EMBED;  // 1M bf16
  __bf16* Xq  = (__bf16*)d_ws;
  __bf16* Xk  = Xq + SZX;
  __bf16* Xv  = Xk + SZX;
  __bf16* Wqb = Xv + SZX;
  __bf16* Wkb = Wqb + SZW;
  __bf16* Wvb = Wkb + SZW;
  __bf16* Wob = Wvb + SZW;
  __bf16* Qh  = Wob + SZW;
  __bf16* Kh  = Qh + SZX;
  __bf16* Vh  = Kh + SZX;
  __bf16* Ctx = Vh + SZX;

  dim3 blk(256);
  int nconv = (int)SZX;
  dim3 gconv((nconv + 255) / 256);
  cvt_f32_to_bf16<<<gconv, blk, 0, stream>>>(q, Xq, nconv);
  cvt_f32_to_bf16<<<gconv, blk, 0, stream>>>(k, Xk, nconv);
  cvt_f32_to_bf16<<<gconv, blk, 0, stream>>>(v, Xv, nconv);
  int nw = (int)SZW;
  dim3 gw((nw + 255) / 256);
  cvt_f32_to_bf16<<<gw, blk, 0, stream>>>(Wq, Wqb, nw);
  cvt_f32_to_bf16<<<gw, blk, 0, stream>>>(Wk, Wkb, nw);
  cvt_f32_to_bf16<<<gw, blk, 0, stream>>>(Wv, Wvb, nw);
  cvt_f32_to_bf16<<<gw, blk, 0, stream>>>(Wo, Wob, nw);

  dim3 ggemm(EMBED / 128, MROWS / 128);
  gemm_bf16_128x128<<<ggemm, blk, 0, stream>>>(Xq, Wqb, bq, Qh, nullptr, MROWS, EMBED, EMBED, 0);
  gemm_bf16_128x128<<<ggemm, blk, 0, stream>>>(Xk, Wkb, bk, Kh, nullptr, MROWS, EMBED, EMBED, 0);
  gemm_bf16_128x128<<<ggemm, blk, 0, stream>>>(Xv, Wvb, bv, Vh, nullptr, MROWS, EMBED, EMBED, 0);

  dim3 gattn(SEQ / 64, HEADS, BB);
  attn_fwd<<<gattn, blk, 0, stream>>>(Qh, Kh, Vh, pm, Ctx);

  gemm_bf16_128x128<<<ggemm, blk, 0, stream>>>(Ctx, Wob, bo, nullptr, out, MROWS, EMBED, EMBED, 1);
}
